// GCN_LSTM_89421219102803
// MI455X (gfx1250) — matrix operand order repaired and checked by reading, not compiled, not run on hardware
//
#include <hip/hip_runtime.h>

typedef __bf16 bf16;
typedef __attribute__((ext_vector_type(16))) __bf16 v16bf;
typedef __attribute__((ext_vector_type(8)))  __bf16 v8bf;
typedef __attribute__((ext_vector_type(8)))  float   v8f;
typedef __attribute__((ext_vector_type(4)))  int     v4i;

union BF16Frag { v16bf v; v8bf h[2]; };

// ---- problem dimensions ----
constexpr int T_   = 512;
constexpr int N_   = 60;
constexpr int F_   = 2;
constexpr int C_   = 64;
constexpr int H_   = 512;
constexpr int NC_  = N_ * C_;        // 3840 : LSTM input width (K of gates GEMM)
constexpr int H4_  = 4 * H_;         // 2048 : gates width
constexpr int TN_  = T_ * N_;        // 30720
constexpr int FRAME_ = TN_ * F_;     // 61440 floats per output frame
constexpr int KP3_ = 1056;           // 1028 padded up to multiple of 32 (K of MLP L3)
constexpr int NP2_ = 1088;           // 1028 padded up to multiple of 64 (N of MLP L2)
constexpr int NP4_ = 128;            // 120 padded up to multiple of 64 (N of MLP L4)

#if __has_builtin(__builtin_amdgcn_global_load_async_to_lds_b128) && \
    __has_builtin(__builtin_amdgcn_s_wait_asynccnt)
#define ASYNC_LDS 1
typedef __attribute__((address_space(1))) v4i gv4i;   // global b128 payload
typedef __attribute__((address_space(3))) v4i lv4i;   // LDS b128 payload
#else
#define ASYNC_LDS 0
#endif

__device__ __forceinline__ float sigmoidf_(float x) { return 1.0f / (1.0f + __expf(-x)); }

// ---- bf16x2 split helpers (hi + lo ~= fp32) ----
__device__ __forceinline__ bf16 split_hi(float v) { return (bf16)v; }
__device__ __forceinline__ bf16 split_lo(float v) { return (bf16)(v - (float)((bf16)v)); }

// ---- WMMA fragment loaders, per CDNA5 ISA §7.12.2 VGPR layouts (wave32) ----
// A (16-bit, 16x32): lane L: M = L%16; VGPR0..3 hold K = (L/16)*8 + 0..7,
//                    VGPR4..7 hold K = 16 + (L/16)*8 + 0..7.
__device__ __forceinline__ v16bf load_a_frag(const bf16* A, int lda, int m0, int lane, int k0) {
  int half = lane >> 4;
  int m = m0 + (lane & 15);
  const bf16* p = A + (size_t)m * lda + k0 + half * 8;
  BF16Frag f;
  f.h[0] = *(const v8bf*)p;          // K = k0 + half*8 .. +7   (16 B contiguous)
  f.h[1] = *(const v8bf*)(p + 16);   // K = k0 + 16 + half*8 .. (16 B contiguous)
  return f.v;
}
// B (16-bit, 32x16): lane L: N = L%16; same K order as A: K = (L/16)*8 + 0..7, then 16 + (L/16)*8 + 0..7.
// B tile staged in LDS as [64 rows][32 k] bf16 -> two 16 B chunks per lane (k = half*8.., 16 + half*8..).
__device__ __forceinline__ v16bf load_b_shared(const bf16* sh, int nt, int lane) {
  int half = lane >> 4;
  const bf16* p = sh + (nt * 16 + (lane & 15)) * 32 + half * 8;
  BF16Frag f;
  f.h[0] = *(const v8bf*)p;
  f.h[1] = *(const v8bf*)(p + 16);
  return f.v;
}

// Stage a 64x32 bf16 tile (4 KB) from B[n0..n0+63][k0..k0+31] (ldb-strided, [N][K])
// into LDS. 128 threads x 2 chunks x 16 B. Async global->LDS when available.
__device__ __forceinline__ void stage_tile(const bf16* __restrict__ B, int ldb, int n0, int k0,
                                           bf16* sh, int tid) {
#pragma unroll
  for (int r = 0; r < 2; ++r) {
    int e = (tid + 128 * r) * 8;        // element index; 8 bf16 per 16 B chunk
    int row = e >> 5, kc = e & 31;
    const bf16* g = B + (size_t)(n0 + row) * ldb + k0 + kc;
#if ASYNC_LDS
    __builtin_amdgcn_global_load_async_to_lds_b128(
        (gv4i*)(uintptr_t)g,                          // global src (addrspace 1)
        (lv4i*)(uint32_t)(uintptr_t)(sh + e),         // LDS dst: addr[31:0] = LDS offset
        0, 0);
#else
    *(v8bf*)(sh + e) = *(const v8bf*)g;
#endif
  }
}

#define WMMA_BF16(a, b, c) \
  __builtin_amdgcn_wmma_f32_16x16x32_bf16(false, (a), false, (b), (short)0, (c), false, false)

// ================= generic split-bf16 WMMA GEMM =================
// C[m,n] = sum_k A1[m,k]*B1[n,k] + sum_k A2[m,k]*B2[n,k] + bias1[n] + bias2[n]
// Workgroup = 4 waves -> 64x64 output tile. B hi/lo tiles double-buffered in LDS via
// async global->LDS; each wave computes a 16x64 strip: 3 WMMAs per K-step per n-tile
// (hi*hi + hi*lo + lo*hi) for ~fp32 accuracy with bf16 matrix-core throughput.
// epi: 0 = store fp32 to C1 (and C2 if non-null); 1 = ReLU then re-split to Oh/Ol.
__global__ __launch_bounds__(128) void wmma_gemm(
    const bf16* __restrict__ A1h, const bf16* __restrict__ A1l,
    const bf16* __restrict__ B1h, const bf16* __restrict__ B1l,
    int K1, int lda1, int ldb1,
    const bf16* __restrict__ A2h, const bf16* __restrict__ A2l,
    const bf16* __restrict__ B2h, const bf16* __restrict__ B2l,
    int K2, int lda2, int ldb2,
    const float* __restrict__ bias1, const float* __restrict__ bias2,
    int Nreal, int epi,
    float* __restrict__ C1, float* __restrict__ C2, int ldc,
    bf16* __restrict__ Oh, bf16* __restrict__ Ol, int ldo)
{
  __shared__ bf16 shB[2][2][64 * 32];   // [buffer][hi/lo][64 rows x 32 k]

  const int tid  = threadIdx.x;
  const int lane = tid & 31;
  const int wv   = tid >> 5;
  const int mb   = blockIdx.y * 64 + wv * 16;   // this wave's 16 M-rows
  const int n0   = blockIdx.x * 64;

  v8f acc[4] = {};

#pragma unroll
  for (int seg = 0; seg < 2; ++seg) {
    const bf16* Ah = (seg == 0) ? A1h : A2h;
    const bf16* Al = (seg == 0) ? A1l : A2l;
    const bf16* Bh = (seg == 0) ? B1h : B2h;
    const bf16* Bl = (seg == 0) ? B1l : B2l;
    const int K   = (seg == 0) ? K1 : K2;
    const int lda = (seg == 0) ? lda1 : lda2;
    const int ldb = (seg == 0) ? ldb1 : ldb2;
    if (K <= 0) continue;

    stage_tile(Bh, ldb, n0, 0, shB[0][0], tid);
    stage_tile(Bl, ldb, n0, 0, shB[0][1], tid);

    for (int k0 = 0; k0 < K; k0 += 32) {
      const int cur = (k0 >> 5) & 1;
      const bool more = (k0 + 32) < K;
      if (more) {                        // prefetch next K-step into other buffer
        stage_tile(Bh, ldb, n0, k0 + 32, shB[cur ^ 1][0], tid);
        stage_tile(Bl, ldb, n0, k0 + 32, shB[cur ^ 1][1], tid);
      }
#if ASYNC_LDS
      // current buffer's 4 async loads (issued one round ago) must be done;
      // allow the 4 just-issued prefetch loads to remain in flight.
      if (more) __builtin_amdgcn_s_wait_asynccnt(4);
      else      __builtin_amdgcn_s_wait_asynccnt(0);
#endif
      __syncthreads();

      v16bf ah = load_a_frag(Ah, lda, mb, lane, k0);
      v16bf al = load_a_frag(Al, lda, mb, lane, k0);
#pragma unroll
      for (int nt = 0; nt < 4; ++nt) {
        v16bf bh = load_b_shared(shB[cur][0], nt, lane);
        v16bf bl = load_b_shared(shB[cur][1], nt, lane);
        acc[nt] = WMMA_BF16(ah, bh, acc[nt]);
        acc[nt] = WMMA_BF16(ah, bl, acc[nt]);
        acc[nt] = WMMA_BF16(al, bh, acc[nt]);
      }
      __syncthreads();                   // buffer may be restaged next iteration
    }
  }

  // Epilogue. C/D layout: VGPR v -> M = mb + v + 8*(lane/16), N = n0 + nt*16 + lane%16.
  const int half = lane >> 4;
  const int nl   = lane & 15;
#pragma unroll
  for (int nt = 0; nt < 4; ++nt) {
    int n = n0 + nt * 16 + nl;
    if (n >= Nreal) continue;
    float b = (bias1 ? bias1[n] : 0.0f) + (bias2 ? bias2[n] : 0.0f);
#pragma unroll
    for (int v = 0; v < 8; ++v) {
      int m = mb + v + 8 * half;
      float val = acc[nt][v] + b;
      if (epi == 1) {
        val = fmaxf(val, 0.0f);
        size_t o = (size_t)m * ldo + n;
        Oh[o] = split_hi(val);
        Ol[o] = split_lo(val);
      } else {
        size_t o = (size_t)m * ldc + n;
        C1[o] = val;
        if (C2) C2[o] = val;
      }
    }
  }
}

// ================= GCN layer (small graphs: N=60, C=64) =================
__global__ __launch_bounds__(256) void gcn_kernel(
    const float* __restrict__ x, const float* __restrict__ adj,
    const float* __restrict__ W, const float* __restrict__ b,
    bf16* __restrict__ outh, bf16* __restrict__ outl)
{
  __shared__ float hsm[N_ * C_];
  __shared__ float nrm[N_];
  __shared__ float xs[N_ * F_];
  const int t = blockIdx.x;
  const float* xt = x + (size_t)t * N_ * F_;
  const float* At = adj + (size_t)t * N_ * N_;

  for (int i = threadIdx.x; i < N_ * F_; i += blockDim.x) xs[i] = xt[i];
  __syncthreads();
  for (int i = threadIdx.x; i < N_ * C_; i += blockDim.x) {
    int j = i / C_, c = i % C_;
    hsm[i] = xs[j * F_ + 0] * W[0 * C_ + c] + xs[j * F_ + 1] * W[1 * C_ + c];
  }
  for (int i = threadIdx.x; i < N_; i += blockDim.x) {
    float d = 1.0f;                       // self-loop from +eye
    for (int j = 0; j < N_; ++j) d += At[i * N_ + j];
    nrm[i] = (d > 0.0f) ? rsqrtf(d) : 0.0f;
  }
  __syncthreads();
  for (int idx = threadIdx.x; idx < N_ * C_; idx += blockDim.x) {
    int i = idx / C_, c = idx % C_;
    float s = 0.0f;
    for (int j = 0; j < N_; ++j) {
      float a = At[i * N_ + j] + (i == j ? 1.0f : 0.0f);
      s += a * nrm[j] * hsm[j * C_ + c];
    }
    float val = nrm[i] * s + b[c];
    size_t o = (size_t)t * NC_ + idx;
    outh[o] = split_hi(val);
    outl[o] = split_lo(val);
  }
}

// ================= LSTM pointwise update =================
__global__ __launch_bounds__(256) void lstm_update(
    const float* __restrict__ gates, float* __restrict__ c,
    bf16* __restrict__ hh, bf16* __restrict__ hl)
{
  int idx = blockIdx.x * blockDim.x + threadIdx.x;
  if (idx >= T_ * H_) return;
  int t = idx / H_, n = idx % H_;
  const float* g = gates + (size_t)t * H4_;
  float ig = sigmoidf_(g[n]);
  float fg = sigmoidf_(g[H_ + n]);
  float gg = tanhf(g[2 * H_ + n]);
  float og = sigmoidf_(g[3 * H_ + n]);
  float cn = fg * c[idx] + ig * gg;
  c[idx] = cn;
  float h = og * tanhf(cn);
  hh[idx] = split_hi(h);
  hl[idx] = split_lo(h);
}

// ================= radius-graph adjacency (decoder) =================
__global__ __launch_bounds__(256) void find_adj_kernel(
    const float* __restrict__ pred, const float* __restrict__ stdv,
    const float* __restrict__ meanv, float* __restrict__ adj)
{
  __shared__ float px[N_], py[N_];
  __shared__ int ex[N_];
  const int t = blockIdx.x;
  const float* p = pred + (size_t)t * N_ * F_;
  if (threadIdx.x < N_) {
    float x0 = p[threadIdx.x * 2 + 0] * stdv[0] + meanv[0];
    float x1 = p[threadIdx.x * 2 + 1] * stdv[1] + meanv[1];
    px[threadIdx.x] = x0; py[threadIdx.x] = x1;
    ex[threadIdx.x] = (x0 > 0.04f) && (x1 > 0.04f);
  }
  __syncthreads();
  for (int ij = threadIdx.x; ij < N_ * N_; ij += blockDim.x) {
    int i = ij / N_, j = ij % N_;
    float dx = px[i] - px[j], dy = py[i] - py[j];
    float dist = sqrtf(dx * dx + dy * dy);
    bool conn = ((dist > 0.0f) && (dist < 10.0f)) || (i == j);
    adj[(size_t)t * N_ * N_ + ij] = (ex[i] && ex[j] && conn) ? 1.0f : 0.0f;
  }
}

// ======== weight convert: fp32 -> bf16 hi/lo, optional transpose, zero-pad ========
__global__ __launch_bounds__(256) void cvt_split(
    const float* __restrict__ src, int srcR, int srcC, int trans,
    bf16* __restrict__ oh, bf16* __restrict__ ol, int On, int Ok)
{
  int idx = blockIdx.x * blockDim.x + threadIdx.x;
  if (idx >= On * Ok) return;
  int n = idx / Ok, k = idx % Ok;
  float v = 0.0f;
  if (!trans) { if (n < srcR && k < srcC) v = src[(size_t)n * srcC + k]; }
  else        { if (k < srcR && n < srcC) v = src[(size_t)k * srcC + n]; }
  oh[idx] = split_hi(v);
  ol[idx] = split_lo(v);
}

__global__ __launch_bounds__(256) void zero_f32(float* p, int n) {
  int i = blockIdx.x * blockDim.x + threadIdx.x;
  if (i < n) p[i] = 0.0f;
}
__global__ __launch_bounds__(256) void zero_bf16(bf16* p, int n) {
  int i = blockIdx.x * blockDim.x + threadIdx.x;
  if (i < n) p[i] = (bf16)0.0f;
}
__global__ __launch_bounds__(256) void copy_f32(float* dst, const float* src, int n) {
  int i = blockIdx.x * blockDim.x + threadIdx.x;
  if (i < n) dst[i] = src[i];
}

// ================= host orchestration =================
extern "C" void kernel_launch(void* const* d_in, const int* in_sizes, int n_in,
                              void* d_out_, int out_size, void* d_ws, size_t ws_size,
                              hipStream_t stream) {
  const float* feature = (const float*)d_in[0];
  const float* adj_in  = (const float*)d_in[1];
  const float* stdv    = (const float*)d_in[2];
  const float* meanv   = (const float*)d_in[3];
  const float* gcn_W   = (const float*)d_in[4];
  const float* gcn_b   = (const float*)d_in[5];
  const float* W_ih    = (const float*)d_in[6];
  const float* W_hh    = (const float*)d_in[7];
  const float* b_ih    = (const float*)d_in[8];
  const float* b_hh    = (const float*)d_in[9];
  const float* W1 = (const float*)d_in[10]; const float* b1 = (const float*)d_in[11];
  const float* W2 = (const float*)d_in[12]; const float* b2 = (const float*)d_in[13];
  const float* W3 = (const float*)d_in[14]; const float* b3 = (const float*)d_in[15];
  const float* W4 = (const float*)d_in[16]; const float* b4 = (const float*)d_in[17];
  float* d_out = (float*)d_out_;

  char* base = (char*)d_ws;
  size_t off = 0;
  auto alloc = [&](size_t bytes) -> char* {
    off = (off + 255) & ~(size_t)255;
    char* p = base + off; off += bytes; return p;
  };

  // weights (bf16 hi/lo, [N][K] layout, padded)
  bf16* Wih_h = (bf16*)alloc((size_t)H4_ * NC_ * 2);  bf16* Wih_l = (bf16*)alloc((size_t)H4_ * NC_ * 2);
  bf16* Whh_h = (bf16*)alloc((size_t)H4_ * H_ * 2);   bf16* Whh_l = (bf16*)alloc((size_t)H4_ * H_ * 2);
  bf16* W1T_h = (bf16*)alloc((size_t)512 * 512 * 2);  bf16* W1T_l = (bf16*)alloc((size_t)512 * 512 * 2);
  bf16* W2T_h = (bf16*)alloc((size_t)NP2_ * 512 * 2); bf16* W2T_l = (bf16*)alloc((size_t)NP2_ * 512 * 2);
  bf16* W3T_h = (bf16*)alloc((size_t)512 * KP3_ * 2); bf16* W3T_l = (bf16*)alloc((size_t)512 * KP3_ * 2);
  bf16* W4T_h = (bf16*)alloc((size_t)NP4_ * 512 * 2); bf16* W4T_l = (bf16*)alloc((size_t)NP4_ * 512 * 2);
  // activations
  bf16* x_h  = (bf16*)alloc((size_t)T_ * NC_ * 2);    bf16* x_l  = (bf16*)alloc((size_t)T_ * NC_ * 2);
  bf16* h_h  = (bf16*)alloc((size_t)T_ * H_ * 2);     bf16* h_l  = (bf16*)alloc((size_t)T_ * H_ * 2);
  float* cst = (float*)alloc((size_t)T_ * H_ * 4);
  float* gts = (float*)alloc((size_t)T_ * H4_ * 4);
  bf16* a1_h = (bf16*)alloc((size_t)T_ * 512 * 2);    bf16* a1_l = (bf16*)alloc((size_t)T_ * 512 * 2);
  bf16* a2_h = (bf16*)alloc((size_t)T_ * KP3_ * 2);   bf16* a2_l = (bf16*)alloc((size_t)T_ * KP3_ * 2);
  bf16* a3_h = (bf16*)alloc((size_t)T_ * 512 * 2);    bf16* a3_l = (bf16*)alloc((size_t)T_ * 512 * 2);
  float* pred = (float*)alloc((size_t)T_ * 120 * 4);
  float* adjw = (float*)alloc((size_t)T_ * N_ * N_ * 4);

  auto blks = [](int n) { return (n + 255) / 256; };

  // ---- weight conversion (every launch; stateless/deterministic) ----
  cvt_split<<<blks(H4_ * NC_), 256, 0, stream>>>(W_ih, H4_, NC_, 0, Wih_h, Wih_l, H4_, NC_);
  cvt_split<<<blks(H4_ * H_), 256, 0, stream>>>(W_hh, H4_, H_, 0, Whh_h, Whh_l, H4_, H_);
  cvt_split<<<blks(512 * 512), 256, 0, stream>>>(W1, 512, 512, 1, W1T_h, W1T_l, 512, 512);
  cvt_split<<<blks(NP2_ * 512), 256, 0, stream>>>(W2, 512, 1028, 1, W2T_h, W2T_l, NP2_, 512);
  cvt_split<<<blks(512 * KP3_), 256, 0, stream>>>(W3, 1028, 512, 1, W3T_h, W3T_l, 512, KP3_);
  cvt_split<<<blks(NP4_ * 512), 256, 0, stream>>>(W4, 512, 120, 1, W4T_h, W4T_l, NP4_, 512);

  // ---- state / pad init ----
  zero_bf16<<<blks(T_ * H_), 256, 0, stream>>>(h_h, T_ * H_);
  zero_bf16<<<blks(T_ * H_), 256, 0, stream>>>(h_l, T_ * H_);
  zero_f32 <<<blks(T_ * H_), 256, 0, stream>>>(cst, T_ * H_);
  zero_bf16<<<blks(T_ * KP3_), 256, 0, stream>>>(a2_h, T_ * KP3_);  // zero K-pad cols for L3
  zero_bf16<<<blks(T_ * KP3_), 256, 0, stream>>>(a2_l, T_ * KP3_);

  // enc[0] = feature_input[0]
  copy_f32<<<blks(FRAME_), 256, 0, stream>>>(d_out, feature, FRAME_);

  auto gemm = [&](const bf16* A1h, const bf16* A1l, const bf16* B1h, const bf16* B1l,
                  int K1, int lda1, int ldb1,
                  const bf16* A2h, const bf16* A2l, const bf16* B2h, const bf16* B2l,
                  int K2, int lda2, int ldb2,
                  const float* bb1, const float* bb2, int Nreal, int epi,
                  float* C1, float* C2, int ldc, bf16* Oh, bf16* Ol, int ldo) {
    dim3 g((Nreal + 63) / 64, T_ / 64);   // workgroup covers 64x64 tile, 4 waves
    wmma_gemm<<<g, 128, 0, stream>>>(A1h, A1l, B1h, B1l, K1, lda1, ldb1,
                                     A2h, A2l, B2h, B2l, K2, lda2, ldb2,
                                     bb1, bb2, Nreal, epi, C1, C2, ldc, Oh, Ol, ldo);
  };

  for (int s = 0; s < 7; ++s) {
    const float* x_src = (s <= 4) ? feature + (size_t)s * FRAME_ : pred;
    const float* a_src;
    if (s <= 4) {
      a_src = adj_in + (size_t)s * T_ * N_ * N_;
    } else {
      find_adj_kernel<<<T_, 256, 0, stream>>>(pred, stdv, meanv, adjw);
      a_src = adjw;
    }
    float* outp = (s < 4) ? d_out + (size_t)(s + 1) * FRAME_
                          : d_out + (size_t)(5 + (s - 4)) * FRAME_;

    // GCN -> x split
    gcn_kernel<<<T_, 256, 0, stream>>>(x_src, a_src, gcn_W, gcn_b, x_h, x_l);
    // gates = x @ W_ih^T + h @ W_hh^T + b_ih + b_hh
    gemm(x_h, x_l, Wih_h, Wih_l, NC_, NC_, NC_,
         h_h, h_l, Whh_h, Whh_l, H_, H_, H_,
         b_ih, b_hh, H4_, 0, gts, nullptr, H4_, nullptr, nullptr, 0);
    // LSTM pointwise
    lstm_update<<<blks(T_ * H_), 256, 0, stream>>>(gts, cst, h_h, h_l);
    // MLP: H->512->1028->512->120
    gemm(h_h, h_l, W1T_h, W1T_l, H_, H_, H_,
         nullptr, nullptr, nullptr, nullptr, 0, 0, 0,
         b1, nullptr, 512, 1, nullptr, nullptr, 0, a1_h, a1_l, 512);
    gemm(a1_h, a1_l, W2T_h, W2T_l, 512, 512, 512,
         nullptr, nullptr, nullptr, nullptr, 0, 0, 0,
         b2, nullptr, 1028, 1, nullptr, nullptr, 0, a2_h, a2_l, KP3_);
    gemm(a2_h, a2_l, W3T_h, W3T_l, KP3_, KP3_, KP3_,
         nullptr, nullptr, nullptr, nullptr, 0, 0, 0,
         b3, nullptr, 512, 1, nullptr, nullptr, 0, a3_h, a3_l, 512);
    gemm(a3_h, a3_l, W4T_h, W4T_l, 512, 512, 512,
         nullptr, nullptr, nullptr, nullptr, 0, 0, 0,
         b4, nullptr, 120, 0, outp, pred, 120, nullptr, nullptr, 0);
  }
}
